// FullAttention_14207751815222
// MI455X (gfx1250) — matrix operand order repaired and checked by reading, not compiled, not run on hardware
//
#include <hip/hip_runtime.h>
#include <hip/hip_bf16.h>
#include <math.h>

typedef __attribute__((ext_vector_type(16))) __bf16 v16bf;
typedef __attribute__((ext_vector_type(8)))  float  v8f;

#define B_   2
#define S_   2048
#define D_   2048
#define H_   16
#define KV_  4
#define HD_  128
#define RD_  64
#define NREP_ 4

// GEMM block tiling
#define MBLK 128
#define NBLK 64
#define KBLK 32
#define APAD 36   // padded LDS row stride (floats) for A tile, 16B-aligned
#define BPAD 68   // padded LDS row stride (floats) for B tile, 16B-aligned

// ---------------------------------------------------------------------------
// helpers
// ---------------------------------------------------------------------------
__device__ __forceinline__ float halfmax16(float v) {
#pragma unroll
  for (int m = 1; m < 16; m <<= 1) v = fmaxf(v, __shfl_xor(v, m, 32));
  return v;
}
__device__ __forceinline__ float halfsum16(float v) {
#pragma unroll
  for (int m = 1; m < 16; m <<= 1) v += __shfl_xor(v, m, 32);
  return v;
}
__device__ __forceinline__ float wavesum32(float v) {
#pragma unroll
  for (int m = 1; m < 32; m <<= 1) v += __shfl_xor(v, m, 32);
  return v;
}
// wave-local LDS ordering (attention kernel: single wave, no block barrier)
__device__ __forceinline__ void wave_lds_sync() {
  asm volatile("s_wait_dscnt 0x0" ::: "memory");
}
// CDNA5 async global->LDS copy (ASYNCcnt-tracked), 16 bytes per lane
__device__ __forceinline__ void async_load_b128(const float* g, const float* l) {
  unsigned lds_off = (unsigned)(uintptr_t)l;   // addr[31:0] is the LDS address
  asm volatile("global_load_async_to_lds_b128 %0, %1, off"
               :: "v"(lds_off), "v"(g) : "memory");
}
__device__ __forceinline__ void async_wait0() {
  asm volatile("s_wait_asynccnt 0x0" ::: "memory");
}

// ---------------------------------------------------------------------------
// Cooperative tile fetch: A 128x32 f32 tile (1024 x b128), B 32x64 f32 tile
// (512 x b128); 256 threads issue 6 async b128 ops each.
// ---------------------------------------------------------------------------
__device__ __forceinline__ void issue_tile(
    const float* __restrict__ A, const float* __restrict__ W,
    int K, int N, int m0, int n0, int k0,
    float* sA, float* sB, int tid)
{
#pragma unroll
  for (int i = 0; i < 4; ++i) {
    const int c = tid + i * 256;
    const int row = c >> 3, seg = c & 7;
    async_load_b128(A + (size_t)(m0 + row) * K + k0 + seg * 4,
                    sA + row * APAD + seg * 4);
  }
#pragma unroll
  for (int i = 0; i < 2; ++i) {
    const int c = tid + i * 256;
    const int kr = c >> 4, seg = c & 15;
    async_load_b128(W + (size_t)(k0 + kr) * N + n0 + seg * 4,
                    sB + kr * BPAD + seg * 4);
  }
}

// ---------------------------------------------------------------------------
// fp32-in/fp32-out GEMM via bf16 WMMA with double-buffered async-LDS staging.
// C[M,N] = A[M,K] @ W[K,N].  Block = 256 threads (8 waves) -> 128x64 tile.
// Wave w owns rows [w*16, w*16+16) x all 64 cols: 4 WMMA per 32-deep K-step.
// grid = (N/64, M/128).  M,N,K multiples of 128/64/32.
// ---------------------------------------------------------------------------
__global__ __launch_bounds__(256) void wmma_gemm_f32(
    const float* __restrict__ A, const float* __restrict__ W,
    float* __restrict__ C, int M, int N, int K)
{
  __shared__ float shA[2][MBLK * APAD];
  __shared__ float shB[2][KBLK * BPAD];
  const int tid  = threadIdx.x;
  const int lane = tid & 31;
  const int wave = tid >> 5;
  const int lr   = lane & 15;
  const int half = lane >> 4;
  const int n0 = blockIdx.x * NBLK;
  const int m0 = blockIdx.y * MBLK;

  const v8f z8 = {0.f,0.f,0.f,0.f,0.f,0.f,0.f,0.f};
  v8f acc[4];
#pragma unroll
  for (int i = 0; i < 4; ++i) acc[i] = z8;

  issue_tile(A, W, K, N, m0, n0, 0, shA[0], shB[0], tid);

  const int nk = K / KBLK;
  for (int kt = 0; kt < nk; ++kt) {
    async_wait0();        // own wave's inflight tile complete
    __syncthreads();      // all waves' portions visible; prev buffer free
    if (kt + 1 < nk)
      issue_tile(A, W, K, N, m0, n0, (kt + 1) * KBLK,
                 shA[(kt + 1) & 1], shB[(kt + 1) & 1], tid);

    const float* As = shA[kt & 1];
    const float* Bs = shB[kt & 1];

    v16bf a;                                       // A frag: 16x32
    const float* ap = As + (wave * 16 + lr) * APAD + 8 * half;
#pragma unroll
    for (int e = 0; e < 8; ++e) { a[e] = (__bf16)ap[e]; a[8+e] = (__bf16)ap[16+e]; }

#pragma unroll
    for (int nt = 0; nt < 4; ++nt) {               // B frags: 32x16 each
      v16bf b;
      const float* bp = Bs + (8 * half) * BPAD + nt * 16 + lr;
#pragma unroll
      for (int e = 0; e < 8; ++e) { b[e] = (__bf16)bp[e * BPAD]; b[8+e] = (__bf16)bp[(16 + e) * BPAD]; }
      acc[nt] = __builtin_amdgcn_wmma_f32_16x16x32_bf16(false, a, false, b,
                                                        (short)0, acc[nt],
                                                        false, false);
    }
  }

  float* crow = C + (size_t)(m0 + wave * 16 + 8 * half) * N + n0 + lr;
#pragma unroll
  for (int nt = 0; nt < 4; ++nt)
#pragma unroll
    for (int r = 0; r < 8; ++r)
      crow[(size_t)r * N + nt * 16] = acc[nt][r];
}

// ---------------------------------------------------------------------------
// Q: RMS-norm + RoPE, in place inside qg (layout (b,s,H,2*HD); only the first
// HD of each 2*HD row is touched, the gate half stays raw).  1 wave per row.
// ---------------------------------------------------------------------------
__global__ __launch_bounds__(256) void qnorm_rope(
    float* __restrict__ qg, const float* __restrict__ cosb,
    const float* __restrict__ sinb, const float* __restrict__ w)
{
  const int lane = threadIdx.x & 31;
  const int wave = threadIdx.x >> 5;
  const int rid  = blockIdx.x * 8 + wave;          // b*S*H + s*H + h
  const int s = (rid >> 4) & (S_ - 1);
  const int b = rid >> 15;
  float* row = qg + (size_t)rid * (2 * HD_);
  const int i0 = lane * 4;

  float4 q = *(const float4*)(row + i0);
  float ss = q.x*q.x + q.y*q.y + q.z*q.z + q.w*q.w;
  ss = wavesum32(ss);
  const float r = rsqrtf(ss * (1.0f / HD_) + 1e-6f);
  float4 w4 = *(const float4*)(w + i0);
  float o0 = q.x*r*w4.x, o1 = q.y*r*w4.y, o2 = q.z*r*w4.z, o3 = q.w*r*w4.w;

  if (lane < 16) {                                  // RoPE on first RD_=64
    const int j0 = i0 ^ 32;
    float4 pq = *(const float4*)(row + j0);
    float4 pw = *(const float4*)(w + j0);
    float p0 = pq.x*r*pw.x, p1 = pq.y*r*pw.y, p2 = pq.z*r*pw.z, p3 = pq.w*r*pw.w;
    const size_t cs = ((size_t)(b * S_ + s)) * RD_ + i0;
    float4 c4 = *(const float4*)(cosb + cs);
    float4 s4 = *(const float4*)(sinb + cs);
    const float sgn = (lane < 8) ? -1.f : 1.f;
    o0 = o0 * c4.x + sgn * p0 * s4.x;
    o1 = o1 * c4.y + sgn * p1 * s4.y;
    o2 = o2 * c4.z + sgn * p2 * s4.z;
    o3 = o3 * c4.w + sgn * p3 * s4.w;
  }
  *(float4*)(row + i0) = make_float4(o0, o1, o2, o3);
}

// ---------------------------------------------------------------------------
// K: RMS-norm + RoPE + transpose (b,s,kv,hd) -> (b,kv,s,hd) into d_out key.
// ---------------------------------------------------------------------------
__global__ __launch_bounds__(256) void knorm_rope(
    const float* __restrict__ kraw, float* __restrict__ kout,
    const float* __restrict__ cosb, const float* __restrict__ sinb,
    const float* __restrict__ w)
{
  const int lane = threadIdx.x & 31;
  const int wave = threadIdx.x >> 5;
  const int rid  = blockIdx.x * 8 + wave;          // b*S*KV + s*KV + kv
  const int kv = rid & (KV_ - 1);
  const int s  = (rid >> 2) & (S_ - 1);
  const int b  = rid >> 13;
  const float* row = kraw + (size_t)rid * HD_;
  const int i0 = lane * 4;

  float4 q = *(const float4*)(row + i0);
  float ss = q.x*q.x + q.y*q.y + q.z*q.z + q.w*q.w;
  ss = wavesum32(ss);
  const float r = rsqrtf(ss * (1.0f / HD_) + 1e-6f);
  float4 w4 = *(const float4*)(w + i0);
  float o0 = q.x*r*w4.x, o1 = q.y*r*w4.y, o2 = q.z*r*w4.z, o3 = q.w*r*w4.w;

  if (lane < 16) {
    const int j0 = i0 ^ 32;
    float4 pq = *(const float4*)(row + j0);
    float4 pw = *(const float4*)(w + j0);
    float p0 = pq.x*r*pw.x, p1 = pq.y*r*pw.y, p2 = pq.z*r*pw.z, p3 = pq.w*r*pw.w;
    const size_t cs = ((size_t)(b * S_ + s)) * RD_ + i0;
    float4 c4 = *(const float4*)(cosb + cs);
    float4 s4 = *(const float4*)(sinb + cs);
    const float sgn = (lane < 8) ? -1.f : 1.f;
    o0 = o0 * c4.x + sgn * p0 * s4.x;
    o1 = o1 * c4.y + sgn * p1 * s4.y;
    o2 = o2 * c4.z + sgn * p2 * s4.z;
    o3 = o3 * c4.w + sgn * p3 * s4.w;
  }
  float* orow = kout + ((size_t)(b * KV_ + kv) * S_ + s) * HD_;
  *(float4*)(orow + i0) = make_float4(o0, o1, o2, o3);
}

// ---------------------------------------------------------------------------
// V transpose (b,s,kv,hd) -> (b,kv,s,hd) into d_out value region.
// ---------------------------------------------------------------------------
__global__ __launch_bounds__(256) void vtrans(
    const float4* __restrict__ vin, float4* __restrict__ vout)
{
  const int total = B_ * S_ * KV_ * (HD_ / 4);
  for (int i = blockIdx.x * blockDim.x + threadIdx.x; i < total;
       i += gridDim.x * blockDim.x) {
    const int hd4 = i & 31;
    const int rid = i >> 5;                        // b*S*KV + s*KV + kv
    const int kv = rid & (KV_ - 1);
    const int s  = (rid >> 2) & (S_ - 1);
    const int b  = rid >> 13;
    vout[((size_t)(b * KV_ + kv) * S_ + s) * 32 + hd4] = vin[i];
  }
}

// ---------------------------------------------------------------------------
// Causal flash attention + fused sigmoid-gate.  1 wave / (b, h, 16 q-rows).
// Key tiles of 32.  QK^T: 8 WMMA, PV: 8 WMMA per tile.  Online softmax with
// half-wave shuffle reductions (C-frag row M = r + 8*half, col N = lane&15).
// ---------------------------------------------------------------------------
__global__ __launch_bounds__(32) void attn_fwd(
    const float* __restrict__ qg, const float* __restrict__ kmat,
    const float* __restrict__ vmat, float* __restrict__ gated)
{
  __shared__ float pb[16 * 32];
  const int lane = threadIdx.x & 31;
  const int lr   = lane & 15;
  const int half = lane >> 4;
  const int s0 = blockIdx.x * 16;
  const int h  = blockIdx.y;
  const int b  = blockIdx.z;
  const int kv = h / NREP_;

  // Q fragments (16 rows x 128), A layout, bf16
  const float* qbase = qg + (((size_t)(b * S_ + s0 + lr)) * H_ + h) * (2 * HD_);
  v16bf qa[4];
#pragma unroll
  for (int kc = 0; kc < 4; ++kc) {
    const float* p = qbase + kc * 32 + 8 * half;
#pragma unroll
    for (int e = 0; e < 8; ++e) { qa[kc][e] = (__bf16)p[e]; qa[kc][8+e] = (__bf16)p[16+e]; }
  }
  const float* kbase = kmat + (size_t)(b * KV_ + kv) * S_ * HD_;
  const float* vbase = vmat + (size_t)(b * KV_ + kv) * S_ * HD_;

  const v8f z8 = {0.f,0.f,0.f,0.f,0.f,0.f,0.f,0.f};
  v8f o[8];
#pragma unroll
  for (int i = 0; i < 8; ++i) o[i] = z8;
  float mrow[8], lrow[8];
#pragma unroll
  for (int r = 0; r < 8; ++r) { mrow[r] = -__builtin_inff(); lrow[r] = 0.f; }

  const float scale = 0.08838834764831845f;        // 128^-0.5

  for (int t0 = 0; t0 <= s0 + 15; t0 += 32) {
    // ---- S = Q @ K^T (16x32 logits) ----
    v8f c0 = z8, c1 = z8;
#pragma unroll
    for (int kc = 0; kc < 4; ++kc) {
      v16bf kb0, kb1;
      const float* kp0 = kbase + (size_t)(t0 + lr) * HD_ + kc * 32 + 8 * half;
      const float* kp1 = kp0 + (size_t)16 * HD_;
#pragma unroll
      for (int e = 0; e < 8; ++e) { kb0[e] = (__bf16)kp0[e]; kb1[e] = (__bf16)kp1[e]; kb0[8+e] = (__bf16)kp0[16+e]; kb1[8+e] = (__bf16)kp1[16+e]; }
      c0 = __builtin_amdgcn_wmma_f32_16x16x32_bf16(false, qa[kc], false, kb0,
                                                   (short)0, c0, false, false);
      c1 = __builtin_amdgcn_wmma_f32_16x16x32_bf16(false, qa[kc], false, kb1,
                                                   (short)0, c1, false, false);
    }
    // ---- online softmax (rows live in 16-lane halves) ----
    float pv0[8], pv1[8];
#pragma unroll
    for (int r = 0; r < 8; ++r) {
      const int sq = s0 + r + 8 * half;
      float l0 = (t0 + lr      <= sq) ? c0[r] * scale : -__builtin_inff();
      float l1 = (t0 + 16 + lr <= sq) ? c1[r] * scale : -__builtin_inff();
      float mnew = fmaxf(mrow[r], halfmax16(fmaxf(l0, l1)));
      float alpha = __expf(mrow[r] - mnew);
      mrow[r] = mnew;
      float p0 = __expf(l0 - mnew);
      float p1 = __expf(l1 - mnew);
      lrow[r] = lrow[r] * alpha + p0 + p1;
      pv0[r] = p0; pv1[r] = p1;
#pragma unroll
      for (int vn = 0; vn < 8; ++vn) o[vn][r] *= alpha;
    }
    // ---- P: C layout -> LDS -> A layout bf16 ----
#pragma unroll
    for (int r = 0; r < 8; ++r) {
      pb[(r + 8 * half) * 32 + lr]      = pv0[r];
      pb[(r + 8 * half) * 32 + 16 + lr] = pv1[r];
    }
    wave_lds_sync();
    v16bf pa;
    {
      const float* pp = pb + lr * 32 + 8 * half;
#pragma unroll
      for (int e = 0; e < 8; ++e) { pa[e] = (__bf16)pp[e]; pa[8+e] = (__bf16)pp[16+e]; }
    }
    wave_lds_sync();
    // ---- O += P @ V ----
#pragma unroll
    for (int vn = 0; vn < 8; ++vn) {
      v16bf vb;
      const float* vp = vbase + (size_t)(t0 + 8 * half) * HD_ + vn * 16 + lr;
#pragma unroll
      for (int e = 0; e < 8; ++e) { vb[e] = (__bf16)vp[(size_t)e * HD_]; vb[8+e] = (__bf16)vp[(size_t)(16 + e) * HD_]; }
      o[vn] = __builtin_amdgcn_wmma_f32_16x16x32_bf16(false, pa, false, vb,
                                                      (short)0, o[vn], false, false);
    }
  }
  // ---- epilogue: normalize, gate, store (b,s,H*HD) ----
#pragma unroll
  for (int r = 0; r < 8; ++r) {
    const float inv = 1.f / halfsum16(lrow[r]);
    const int sq = s0 + r + 8 * half;
    const float* grow = qg + (((size_t)(b * S_ + sq)) * H_ + h) * (2 * HD_) + HD_;
    float* orow = gated + ((size_t)(b * S_ + sq)) * (H_ * HD_) + h * HD_;
#pragma unroll
    for (int vn = 0; vn < 8; ++vn) {
      const int hd = vn * 16 + lr;
      const float g = grow[hd];
      orow[hd] = o[vn][r] * inv * (1.f / (1.f + __expf(-g)));
    }
  }
}

// ---------------------------------------------------------------------------
extern "C" void kernel_launch(void* const* d_in, const int* in_sizes, int n_in,
                              void* d_out, int out_size, void* d_ws, size_t ws_size,
                              hipStream_t stream)
{
  (void)in_sizes; (void)n_in; (void)out_size; (void)ws_size;
  const float* x    = (const float*)d_in[0];
  const float* cosb = (const float*)d_in[1];
  const float* sinb = (const float*)d_in[2];
  const float* wq   = (const float*)d_in[3];
  const float* wk   = (const float*)d_in[4];
  const float* wv   = (const float*)d_in[5];
  const float* wo   = (const float*)d_in[6];
  const float* qnw  = (const float*)d_in[7];
  const float* knw  = (const float*)d_in[8];

  float* out     = (float*)d_out;                               // (B,S,D)
  float* key_out = out + (size_t)B_ * S_ * D_;                  // (B,KV,S,HD)
  float* val_out = key_out + (size_t)B_ * KV_ * S_ * HD_;       // (B,KV,S,HD)

  float* qg    = (float*)d_ws;                                  // (B,S,H,2*HD)
  float* k_raw = qg    + (size_t)B_ * S_ * H_ * 2 * HD_;        // (B,S,KV,HD)
  float* v_raw = k_raw + (size_t)B_ * S_ * KV_ * HD_;
  float* gated = v_raw + (size_t)B_ * S_ * KV_ * HD_;           // (B,S,H*HD)

  const int M = B_ * S_;                                        // 4096
  dim3 blk(256);

  // projections (async-LDS double-buffered bf16-WMMA GEMM)
  wmma_gemm_f32<<<dim3((H_*2*HD_)/NBLK, M/MBLK), blk, 0, stream>>>(x, wq, qg,   M, H_*2*HD_, D_);
  wmma_gemm_f32<<<dim3((KV_*HD_)/NBLK,  M/MBLK), blk, 0, stream>>>(x, wk, k_raw, M, KV_*HD_,  D_);
  wmma_gemm_f32<<<dim3((KV_*HD_)/NBLK,  M/MBLK), blk, 0, stream>>>(x, wv, v_raw, M, KV_*HD_,  D_);

  // norms / rope / transposes
  qnorm_rope<<<(B_*S_*H_)/8,  blk, 0, stream>>>(qg, cosb, sinb, qnw);
  knorm_rope<<<(B_*S_*KV_)/8, blk, 0, stream>>>(k_raw, key_out, cosb, sinb, knw);
  vtrans<<<1024, blk, 0, stream>>>((const float4*)v_raw, (float4*)val_out);

  // causal flash attention + gate
  attn_fwd<<<dim3(S_/16, H_, B_), dim3(32), 0, stream>>>(qg, key_out, val_out, gated);

  // output projection
  wmma_gemm_f32<<<dim3(D_/NBLK, M/MBLK), blk, 0, stream>>>(gated, wo, out, M, D_, D_);
}
